// FlashAttention_13503377178804
// MI455X (gfx1250) — compile-verified, matrix operand order checked by reading, not run on hardware
//
#include <hip/hip_runtime.h>
#include <hip/hip_bf16.h>
#include <math.h>
#include <stdint.h>

// Problem sizes (fixed by the reference)
#define N_TOK 4096
#define DIM   1024

// GEMM tiling: 256 threads = 8 wave32s, block tile 128x128, K-step 32.
// Wave grid 4x2; each wave owns a 32x64 patch = 2x4 16x16 WMMA tiles.
#define BM 128
#define BN 128
#define KC 32
#define LDH 40          // LDS row pitch in halves (32 + 8 pad -> 80B, conflict-free)
#define THREADS 256

typedef __attribute__((ext_vector_type(8)))  float        v8f;
typedef __attribute__((ext_vector_type(8)))  __bf16       v8bf;
typedef __attribute__((ext_vector_type(16))) __bf16       v16bf;
typedef __attribute__((ext_vector_type(4)))  unsigned int v4u;
typedef __attribute__((ext_vector_type(4)))  int          v4i;
typedef __attribute__((ext_vector_type(8)))  int          v8i_t;

__device__ __forceinline__ unsigned short f2bf(float f) {
  unsigned int u = __float_as_uint(f);
  u += 0x7fffu + ((u >> 16) & 1u);      // round-to-nearest-even
  return (unsigned short)(u >> 16);
}

// Pack two f32 -> two bf16 in one dword via v_perm_b32.
__device__ __forceinline__ unsigned int pack2bf(float a, float b) {
  unsigned int ua = __float_as_uint(a);
  unsigned int ub = __float_as_uint(b);
  ua += 0x7fffu + ((ua >> 16) & 1u);
  ub += 0x7fffu + ((ub >> 16) & 1u);
  return __builtin_amdgcn_perm(ub, ua, 0x07060302u);  // {ub[31:16], ua[31:16]}
}

// ---- Async LDS staging (CDNA5 global_load_async_to_lds_b128) ----------------
// Tile: 128 rows x 32 halves, LDS pitch LDH halves. Thread t moves 32 bytes of
// one row via two async b128 transfers (GVS mode). 2 ASYNC ops per wave.
__device__ __forceinline__ void stage_async_bf16(const unsigned short* __restrict__ g, int ld,
                                                 int rowBase, int k0,
                                                 unsigned short* sm, int t) {
  int r = t >> 1, c = (t & 1) << 4;
  unsigned voff = ((unsigned)(rowBase + r) * (unsigned)ld + (unsigned)(k0 + c)) * 2u;
  unsigned lds  = (unsigned)(uintptr_t)(sm + r * LDH + c);   // low 32 bits = LDS offset
  unsigned long long base = (unsigned long long)g;
  asm volatile("global_load_async_to_lds_b128 %0, %1, %2\n\t"
               "global_load_async_to_lds_b128 %3, %4, %2"
               :: "v"(lds), "v"(voff), "s"(base),
                  "v"(lds + 16u), "v"(voff + 16u)
               : "memory");
}

// ---- TDM staging: one tensor_load_to_lds moves a whole 128x32(half) B tile --
// D# per CDNA5 ISA ch.8: 2D tile, data_size=2B, LDS padding 4 DWORDs every
// 16 DWORDs -> 80B pitch (matches LDH). Issued by one wave; TENSORcnt-tracked.
__device__ __forceinline__ void stage_tdm_b(const unsigned short* __restrict__ g,
                                            unsigned ld, unsigned rows,
                                            int nBase, int k0, unsigned short* sm) {
  unsigned long long ga = (unsigned long long)(g + (size_t)nBase * ld + k0);
  unsigned lds = (unsigned)(uintptr_t)sm;

  v4u g0;
  g0[0] = 1u;                                   // count=1 (valid), user mode
  g0[1] = lds;                                  // lds_addr
  g0[2] = (unsigned)(ga & 0xffffffffu);         // global_addr[31:0]
  g0[3] = (unsigned)((ga >> 32) & 0x01ffffffu)  // global_addr[56:32]
          | (2u << 30);                         // type = 2 ("image")

  v8i_t g1;
  g1[0] = (int)((1u << 16)      // data_size = 1 -> 2 bytes
              | (1u << 20)      // pad_enable
              | (3u << 22)      // pad_interval: 16 DWORDs
              | (3u << 25));    // pad_amount:   4 DWORDs
  g1[1] = (int)((ld & 0xffffu) << 16);                          // tensor_dim0[15:0]
  g1[2] = (int)(((ld >> 16) & 0xffffu) | ((rows & 0xffffu) << 16)); // dim0 hi | dim1 lo
  g1[3] = (int)(((rows >> 16) & 0xffffu) | (32u << 16));        // dim1 hi | tile_dim0=32
  g1[4] = 128;                                                  // tile_dim1=128, tile_dim2=0
  g1[5] = (int)ld;                                              // tensor_dim0_stride[31:0]
  g1[6] = 0;                                                    // stride hi | dim1_stride lo
  g1[7] = 0;

  v4i z4 = {0, 0, 0, 0};
  v8i_t z8 = {0, 0, 0, 0, 0, 0, 0, 0};
  __builtin_amdgcn_tensor_load_to_lds(g0, g1, z4, z4, z8, 0);
}

// ---- VALU staging (f32 -> bf16 convert) -------------------------------------
__device__ __forceinline__ void stageA_f32(const float* __restrict__ g, int ld,
                                           int rowBase, int k0, unsigned short* sm, int t) {
  int r = t >> 1, c = (t & 1) << 4;
  const float4* s = reinterpret_cast<const float4*>(g + (size_t)(rowBase + r) * ld + k0 + c);
  unsigned int pk[8];
#pragma unroll
  for (int j = 0; j < 4; ++j) {
    float4 f = s[j];
    pk[2 * j]     = pack2bf(f.x, f.y);
    pk[2 * j + 1] = pack2bf(f.z, f.w);
  }
  uint4* d = reinterpret_cast<uint4*>(sm + r * LDH + c);
  d[0] = make_uint4(pk[0], pk[1], pk[2], pk[3]);
  d[1] = make_uint4(pk[4], pk[5], pk[6], pk[7]);
}

// B tile from row-major f32 weight W[K][Nn]: transpose+convert into LDS.
__device__ __forceinline__ void stageB_wT(const float* __restrict__ g, int ldn,
                                          int nBase, int k0, unsigned short* sm, int t) {
  int n = t & 127, kk = (t >> 7) << 4;
  unsigned int pk[8];
#pragma unroll
  for (int j = 0; j < 16; j += 2) {
    float a = g[(size_t)(k0 + kk + j) * ldn + nBase + n];
    float b = g[(size_t)(k0 + kk + j + 1) * ldn + nBase + n];
    pk[j >> 1] = pack2bf(a, b);
  }
  uint4* d = reinterpret_cast<uint4*>(sm + n * LDH + kk);
  d[0] = make_uint4(pk[0], pk[1], pk[2], pk[3]);
  d[1] = make_uint4(pk[4], pk[5], pk[6], pk[7]);
}

// ---- WMMA fragment load (16-bit A/B layout per CDNA5 ISA 7.12.2) ------------
__device__ __forceinline__ v16bf load_frag(const unsigned short* sm, int row, int lane) {
  int kb = (lane & 16) ? 8 : 0;
  v8bf lo = *reinterpret_cast<const v8bf*>(sm + row * LDH + kb);
  v8bf hi = *reinterpret_cast<const v8bf*>(sm + row * LDH + kb + 16);
  return __builtin_shufflevector(lo, hi, 0, 1, 2, 3, 4, 5, 6, 7,
                                 8, 9, 10, 11, 12, 13, 14, 15);
}

#define WMMA_BF16(a, b, c) \
  __builtin_amdgcn_wmma_f32_16x16x32_bf16(false, (a), false, (b), (short)0, (c), false, false)

// One K-step of WMMA work for this wave (2 A-frags x 4 B-frags).
__device__ __forceinline__ void compute_step(const unsigned short* Asm, const unsigned short* Bsm,
                                             int wm, int wn, int lane, v8f acc[2][4]) {
  const int ar = lane & 15;
  v16bf a0 = load_frag(Asm, wm + ar, lane);
  v16bf a1 = load_frag(Asm, wm + 16 + ar, lane);
  v16bf b0 = load_frag(Bsm, wn + ar, lane);
  v16bf b1 = load_frag(Bsm, wn + 16 + ar, lane);
  v16bf b2 = load_frag(Bsm, wn + 32 + ar, lane);
  v16bf b3 = load_frag(Bsm, wn + 48 + ar, lane);
  acc[0][0] = WMMA_BF16(a0, b0, acc[0][0]);
  acc[0][1] = WMMA_BF16(a0, b1, acc[0][1]);
  acc[0][2] = WMMA_BF16(a0, b2, acc[0][2]);
  acc[0][3] = WMMA_BF16(a0, b3, acc[0][3]);
  acc[1][0] = WMMA_BF16(a1, b0, acc[1][0]);
  acc[1][1] = WMMA_BF16(a1, b1, acc[1][1]);
  acc[1][2] = WMMA_BF16(a1, b2, acc[1][2]);
  acc[1][3] = WMMA_BF16(a1, b3, acc[1][3]);
}

// ---- Async/TDM double-buffered GEMM core (bf16 A and B) ---------------------
// A staged via global_load_async_to_lds_b128 (2 per wave per chunk),
// B staged via one TDM tensor_load_to_lds issued by wave 0.
// Chunk i+1 transfers are issued before waiting on chunk i (latency hiding).
__device__ __forceinline__ void gemm_core_async(const unsigned short* Ag, int lda,
                                                const unsigned short* Bg, int ldb, unsigned rowsB,
                                                int rowBase, int nBase, int Kdim,
                                                unsigned short* Asm0, unsigned short* Bsm0,
                                                unsigned short* Asm1, unsigned short* Bsm1,
                                                v8f acc[2][4]) {
  const int t = threadIdx.x, lane = t & 31, w = t >> 5;
  const int wm = (w >> 1) << 5, wn = (w & 1) << 6;
  const int steps = Kdim / KC;

  stage_async_bf16(Ag, lda, rowBase, 0, Asm0, t);
  if (w == 0) stage_tdm_b(Bg, (unsigned)ldb, rowsB, nBase, 0, Bsm0);

  for (int i = 0; i < steps; ++i) {
    unsigned short* Acur = (i & 1) ? Asm1 : Asm0;
    unsigned short* Bcur = (i & 1) ? Bsm1 : Bsm0;
    unsigned short* Anxt = (i & 1) ? Asm0 : Asm1;
    unsigned short* Bnxt = (i & 1) ? Bsm0 : Bsm1;
    if (i + 1 < steps) {
      stage_async_bf16(Ag, lda, rowBase, (i + 1) * KC, Anxt, t);
      if (w == 0) stage_tdm_b(Bg, (unsigned)ldb, rowsB, nBase, (i + 1) * KC, Bnxt);
      asm volatile("s_wait_asynccnt 2" ::: "memory");      // chunk i landed, i+1 in flight
      if (w == 0) __builtin_amdgcn_s_wait_tensorcnt(1);
    } else {
      asm volatile("s_wait_asynccnt 0" ::: "memory");
      if (w == 0) __builtin_amdgcn_s_wait_tensorcnt(0);
    }
    __syncthreads();                 // all waves' chunk-i staging visible
    compute_step(Acur, Bcur, wm, wn, lane, acc);
    __syncthreads();                 // reads done before buffer is re-staged
  }
}

// ---- Convert-path GEMM core (f32 A, transposed f32 B), single buffer --------
__device__ __forceinline__ void gemm_core_conv(const float* Ag, int lda,
                                               const float* Bg, int ldb,
                                               int rowBase, int nBase,
                                               int Kdim, int kOffB,
                                               unsigned short* Asm, unsigned short* Bsm,
                                               v8f acc[2][4]) {
  const int t = threadIdx.x, lane = t & 31, w = t >> 5;
  const int wm = (w >> 1) << 5, wn = (w & 1) << 6;

  for (int k0 = 0; k0 < Kdim; k0 += KC) {
    stageA_f32(Ag, lda, rowBase, k0, Asm, t);
    stageB_wT(Bg, ldb, nBase, k0 + kOffB, Bsm, t);
    if (k0 + KC < Kdim) {
      int pr = t >> 1;
      __builtin_prefetch(Ag + (size_t)(rowBase + pr) * lda + k0 + KC);
    }
    __syncthreads();
    compute_step(Asm, Bsm, wm, wn, lane, acc);
    __syncthreads();
  }
}

#define ACC_INIT                                              \
  v8f acc[2][4];                                              \
  {                                                           \
    v8f z = {0.f, 0.f, 0.f, 0.f, 0.f, 0.f, 0.f, 0.f};        \
    _Pragma("unroll") for (int i = 0; i < 2; ++i)             \
      _Pragma("unroll") for (int j = 0; j < 4; ++j) acc[i][j] = z; \
  }

// C/D layout: element r of v8f -> row = r + (lane>=16 ? 8 : 0), col = lane&15.

// ---- Kernel 1: QKV projections (z selects Q/K/V) ---------------------------
__global__ void __launch_bounds__(THREADS)
k_qkv(const float* __restrict__ x,
      const float* __restrict__ Wq, const float* __restrict__ bq,
      const float* __restrict__ Wk, const float* __restrict__ bk,
      const float* __restrict__ Wv, const float* __restrict__ bv,
      unsigned short* __restrict__ Qb, unsigned short* __restrict__ Kb,
      unsigned short* __restrict__ Vt) {
  __shared__ unsigned short Asm[BM * LDH];
  __shared__ unsigned short Bsm[BN * LDH];
  const int mode = blockIdx.z;
  const float* W    = (mode == 0) ? Wq : (mode == 1) ? Wk : Wv;
  const float* bias = (mode == 0) ? bq : (mode == 1) ? bk : bv;
  const int rowBase = blockIdx.x * BM;
  const int nBase   = blockIdx.y * BN;

  ACC_INIT;
  gemm_core_conv(x, DIM, W, DIM, rowBase, nBase, DIM, 0, Asm, Bsm, acc);

  const int lane = threadIdx.x & 31, w = threadIdx.x >> 5;
  const int wm = (w >> 1) << 5, wn = (w & 1) << 6;
  const int hi = (lane >> 4) << 3, cn = lane & 15;
#pragma unroll
  for (int tm = 0; tm < 2; ++tm)
#pragma unroll
    for (int tn = 0; tn < 4; ++tn) {
      int colG = nBase + wn + tn * 16 + cn;
      float bb = bias[colG];
#pragma unroll
      for (int r = 0; r < 8; ++r) {
        int rowG = rowBase + wm + tm * 16 + r + hi;
        unsigned short h = f2bf(acc[tm][tn][r] + bb);
        if (mode == 0)      Qb[(size_t)rowG * DIM + colG] = h;
        else if (mode == 1) Kb[(size_t)rowG * DIM + colG] = h;
        else                Vt[(size_t)colG * N_TOK + rowG] = h;  // V transposed
      }
    }
}

// ---- Kernel 2: P = exp(Q K^T / 32) (bf16) + row sums -----------------------
__global__ void __launch_bounds__(THREADS)
k_scores(const unsigned short* __restrict__ Qb, const unsigned short* __restrict__ Kb,
         unsigned short* __restrict__ P, float* __restrict__ lsum) {
  __shared__ unsigned short Asm0[BM * LDH];
  __shared__ unsigned short Bsm0[BN * LDH];
  __shared__ unsigned short Asm1[BM * LDH];
  __shared__ unsigned short Bsm1[BN * LDH];
  const int rowBase = blockIdx.x * BM;
  const int nBase   = blockIdx.y * BN;

  ACC_INIT;
  gemm_core_async(Qb, DIM, Kb, DIM, N_TOK, rowBase, nBase, DIM,
                  Asm0, Bsm0, Asm1, Bsm1, acc);

  const int lane = threadIdx.x & 31, w = threadIdx.x >> 5;
  const int wm = (w >> 1) << 5, wn = (w & 1) << 6;
  const int hi = (lane >> 4) << 3, cn = lane & 15;
  const float sc = 0.03125f;  // 1/sqrt(1024)
#pragma unroll
  for (int tm = 0; tm < 2; ++tm)
#pragma unroll
    for (int r = 0; r < 8; ++r) {
      int rowG = rowBase + wm + tm * 16 + r + hi;
      float s = 0.0f;
#pragma unroll
      for (int tn = 0; tn < 4; ++tn) {
        int colG = nBase + wn + tn * 16 + cn;
        float p = __expf(acc[tm][tn][r] * sc);
        P[(size_t)rowG * N_TOK + colG] = f2bf(p);
        s += p;
      }
      s += __shfl_xor(s, 1);   // reduce the 16 lanes holding this row
      s += __shfl_xor(s, 2);
      s += __shfl_xor(s, 4);
      s += __shfl_xor(s, 8);
      if (cn == 0) atomicAdd(&lsum[rowG], s);
    }
}

// ---- Kernel 3: O = (P @ V) / rowsum ----------------------------------------
__global__ void __launch_bounds__(THREADS)
k_attnv(const unsigned short* __restrict__ P, const unsigned short* __restrict__ Vt,
        const float* __restrict__ lsum, float* __restrict__ O) {
  __shared__ unsigned short Asm0[BM * LDH];
  __shared__ unsigned short Bsm0[BN * LDH];
  __shared__ unsigned short Asm1[BM * LDH];
  __shared__ unsigned short Bsm1[BN * LDH];
  const int rowBase = blockIdx.x * BM;
  const int nBase   = blockIdx.y * BN;

  ACC_INIT;
  gemm_core_async(P, N_TOK, Vt, N_TOK, DIM, rowBase, nBase, N_TOK,
                  Asm0, Bsm0, Asm1, Bsm1, acc);

  const int lane = threadIdx.x & 31, w = threadIdx.x >> 5;
  const int wm = (w >> 1) << 5, wn = (w & 1) << 6;
  const int hi = (lane >> 4) << 3, cn = lane & 15;
#pragma unroll
  for (int tm = 0; tm < 2; ++tm)
#pragma unroll
    for (int r = 0; r < 8; ++r) {
      int rowG = rowBase + wm + tm * 16 + r + hi;
      float inv = 1.0f / lsum[rowG];
#pragma unroll
      for (int tn = 0; tn < 4; ++tn) {
        int colG = nBase + wn + tn * 16 + cn;
        O[(size_t)rowG * DIM + colG] = acc[tm][tn][r] * inv;
      }
    }
}

// ---- Kernel 4: out = sigmoid([x|O] @ Wg + bg) * O --------------------------
__global__ void __launch_bounds__(THREADS)
k_gate(const float* __restrict__ x, const float* __restrict__ O,
       const float* __restrict__ Wg, const float* __restrict__ bg,
       float* __restrict__ out) {
  __shared__ unsigned short Asm[BM * LDH];
  __shared__ unsigned short Bsm[BN * LDH];
  const int rowBase = blockIdx.x * BM;
  const int nBase   = blockIdx.y * BN;

  ACC_INIT;
  // K split: first 1024 from x, next 1024 from O; Wg rows continue 1024..2047.
  gemm_core_conv(x, DIM, Wg, DIM, rowBase, nBase, DIM, 0,   Asm, Bsm, acc);
  gemm_core_conv(O, DIM, Wg, DIM, rowBase, nBase, DIM, DIM, Asm, Bsm, acc);

  const int lane = threadIdx.x & 31, w = threadIdx.x >> 5;
  const int wm = (w >> 1) << 5, wn = (w & 1) << 6;
  const int hi = (lane >> 4) << 3, cn = lane & 15;
#pragma unroll
  for (int tm = 0; tm < 2; ++tm)
#pragma unroll
    for (int tn = 0; tn < 4; ++tn) {
      int colG = nBase + wn + tn * 16 + cn;
      float bb = bg[colG];
#pragma unroll
      for (int r = 0; r < 8; ++r) {
        int rowG = rowBase + wm + tm * 16 + r + hi;
        float z = acc[tm][tn][r] + bb;
        float g = 1.0f / (1.0f + __expf(-z));
        size_t idx = (size_t)rowG * DIM + colG;
        out[idx] = g * O[idx];
      }
    }
}

__global__ void k_zero(float* p, int n) {
  int i = blockIdx.x * blockDim.x + threadIdx.x;
  if (i < n) p[i] = 0.0f;
}

// ---- Launch -----------------------------------------------------------------
extern "C" void kernel_launch(void* const* d_in, const int* in_sizes, int n_in,
                              void* d_out, int out_size, void* d_ws, size_t ws_size,
                              hipStream_t stream) {
  const float* x  = (const float*)d_in[0];
  const float* Wq = (const float*)d_in[1];
  const float* bq = (const float*)d_in[2];
  const float* Wk = (const float*)d_in[3];
  const float* bk = (const float*)d_in[4];
  const float* Wv = (const float*)d_in[5];
  const float* bv = (const float*)d_in[6];
  const float* Wg = (const float*)d_in[7];
  const float* bg = (const float*)d_in[8];
  float* out = (float*)d_out;

  char* ws = (char*)d_ws;
  size_t off = 0;
  unsigned short* Qb = (unsigned short*)(ws + off); off += (size_t)N_TOK * DIM * 2;   // 8 MB
  unsigned short* Kb = (unsigned short*)(ws + off); off += (size_t)N_TOK * DIM * 2;   // 8 MB
  unsigned short* Vt = (unsigned short*)(ws + off); off += (size_t)DIM * N_TOK * 2;   // 8 MB
  unsigned short* P  = (unsigned short*)(ws + off); off += (size_t)N_TOK * N_TOK * 2; // 32 MB
  float* O    = (float*)(ws + off); off += (size_t)N_TOK * DIM * 4;                   // 16 MB
  float* lsum = (float*)(ws + off); off += (size_t)N_TOK * 4;

  k_zero<<<dim3((N_TOK + 255) / 256), dim3(256), 0, stream>>>(lsum, N_TOK);
  k_qkv   <<<dim3(N_TOK / BM, DIM / BN, 3), dim3(THREADS), 0, stream>>>(
      x, Wq, bq, Wk, bk, Wv, bv, Qb, Kb, Vt);
  k_scores<<<dim3(N_TOK / BM, N_TOK / BN), dim3(THREADS), 0, stream>>>(Qb, Kb, P, lsum);
  k_attnv <<<dim3(N_TOK / BM, DIM / BN),   dim3(THREADS), 0, stream>>>(P, Vt, lsum, O);
  k_gate  <<<dim3(N_TOK / BM, DIM / BN),   dim3(THREADS), 0, stream>>>(x, O, Wg, bg, out);
}
